// AlexNet_1W1A_61083024883802
// MI455X (gfx1250) — 8-bit matrix operand order repaired and checked by reading, compiled, not run on hardware
//
#include <hip/hip_runtime.h>
#include <stdint.h>


typedef __attribute__((ext_vector_type(8))) int v8i;
typedef __attribute__((ext_vector_type(4))) int v4i;
typedef __attribute__((ext_vector_type(2))) int v2i;
typedef signed char i8;

#define BN_EPS 1e-4f
#define BATCH  1024

__device__ __forceinline__ float opaque_f(float x) {
  asm volatile("" : "+v"(x));
  return x;
}

__device__ __forceinline__ i8 binact_q(float y) {
  float r = opaque_f(fmaxf(y, 0.0f));
  return (r >= 0.0f) ? (i8)1 : (i8)(-1);
}

__device__ __forceinline__ i8 i8max(i8 a, i8 b) { return a > b ? a : b; }

__global__ void k_prep_wconv(const float* __restrict__ w, i8* __restrict__ wp,
                             int Cout, int Cin, int KPad) {
  int gid = blockIdx.x * 256 + threadIdx.x;
  int total = Cout * KPad;
  if (gid >= total) return;
  int k  = gid % KPad;
  int co = gid / KPad;
  int tap = k >> 7;
  int ci  = k & 127;
  i8 q = 0;
  if (ci < Cin) {
    float wv = w[(co * Cin + ci) * 9 + tap];
    q = (wv >= 0.0f) ? (i8)1 : (i8)(-1);
  }
  wp[gid] = q;
}

__global__ void k_prep_wfc(const float* __restrict__ w, i8* __restrict__ wp,
                           int NfPad, int Nf, int KPad, int K) {
  int gid = blockIdx.x * 256 + threadIdx.x;
  int total = NfPad * KPad;
  if (gid >= total) return;
  int k  = gid % KPad;
  int of = gid / KPad;
  i8 q = 0;
  if (of < Nf && k < K)
    q = (w[of * K + k] >= 0.0f) ? (i8)1 : (i8)(-1);
  wp[gid] = q;
}

__global__ void k_prep_bn(const float* __restrict__ g, const float* __restrict__ b,
                          const float* __restrict__ m, const float* __restrict__ v,
                          const float* __restrict__ bias, int n,
                          float* __restrict__ s_out, float* __restrict__ t_out) {
  int i = blockIdx.x * 256 + threadIdx.x;
  if (i >= n) return;
  float s = g[i] * rsqrtf(v[i] + BN_EPS);
  float t = b[i] - m[i] * s;
  if (bias) t += bias[i] * s;
  s_out[i] = s;
  t_out[i] = t;
}

__global__ void k_conv1(const float* __restrict__ x, const float* __restrict__ w,
                        const float* __restrict__ sc, const float* __restrict__ tt,
                        i8* __restrict__ out) {
  const int C = 96, H = 28, W = 28, Hp = 30, Wp = 30;
  int gid = blockIdx.x * 256 + threadIdx.x;
  int total = BATCH * H * W * C;
  if (gid >= total) return;
  int co = gid % C;
  int p  = gid / C;
  int xc = p % W;
  int t  = p / W;
  int y  = t % H;
  int nb = t / H;
  float acc = 0.0f;
  for (int ky = 0; ky < 3; ++ky) {
    int yy = y + ky - 1;
    if ((unsigned)yy >= (unsigned)H) continue;
    for (int kx = 0; kx < 3; ++kx) {
      int xx = xc + kx - 1;
      if ((unsigned)xx >= (unsigned)W) continue;
      float ws = (w[co * 9 + ky * 3 + kx] >= 0.0f) ? 1.0f : -1.0f;
      acc += ws * x[((size_t)nb * H + yy) * W + xx];
    }
  }
  float yv = acc * sc[co] + tt[co];
  out[((size_t)(nb * Hp + y + 1) * Wp + (xc + 1)) * C + co] = binact_q(yv);
}

template <int Cin, int H, int W, int Cout, int OPAD>
__global__ void __launch_bounds__(256)
k_bconv_wmma(const i8* __restrict__ act, const i8* __restrict__ wp,
             const float* __restrict__ sc, const float* __restrict__ tt,
             i8* __restrict__ out) {
  constexpr int MT   = Cout / 16;
  constexpr int NPIX = BATCH * H * W;
  constexpr int NT   = NPIX / 32;
  constexpr int Hp   = H + 2, Wp = W + 2;
  constexpr int Ho   = H + 2 * OPAD, Wo = W + 2 * OPAD;
  constexpr int KPAD = 1152;

  int wave = (blockIdx.x * 256 + threadIdx.x) >> 5;
  int lane = threadIdx.x & 31;
  if (wave >= MT * NT) return;

  int mt = wave % MT;
  int nt = wave / MT;
  bool hi  = lane >= 16;
  int  l15 = lane & 15;

  int p0 = nt * 32 + l15;
  int p1 = p0 + 16;
  int x0 = p0 % W; int t0 = p0 / W; int y0 = t0 % H; int n0 = t0 / H;
  int x1 = p1 % W; int t1 = p1 / W; int y1 = t1 % H; int n1 = t1 / H;

  const i8* ib0 = act + (size_t)((n0 * Hp + y0) * Wp + x0) * Cin;
  const i8* ib1 = act + (size_t)((n1 * Hp + y1) * Wp + x1) * Cin;
  const i8* wrow = wp + (size_t)(mt * 16 + l15) * KPAD;
  const int aoff = hi ? 16 : 0;
  const int boff = hi ? 16 : 0;

  v8i acc0 = (v8i)0;
  v8i acc1 = (v8i)0;

#pragma unroll
  for (int c = 0; c < 18; ++c) {
    const int k0 = c * 64;
    v2i a0 = *(const v2i*)(wrow + k0 + aoff + 0);
    v2i a1 = *(const v2i*)(wrow + k0 + aoff + 8);
    v2i a2 = *(const v2i*)(wrow + k0 + aoff + 32);
    v2i a3 = *(const v2i*)(wrow + k0 + aoff + 40);
    v8i A;
    A[0] = a0.x; A[1] = a0.y; A[2] = a1.x; A[3] = a1.y;
    A[4] = a2.x; A[5] = a2.y; A[6] = a3.x; A[7] = a3.y;

    const int tap  = c >> 1;
    const int half = (c & 1) << 6;
    const size_t poff = (size_t)((tap / 3) * Wp + (tap % 3)) * Cin;
    const int k1 = half + boff;
    const int k2 = half + 32 + boff;

    v4i b00 = (k1 < Cin) ? *(const v4i*)(ib0 + poff + k1) : (v4i)0;
    v4i b01 = (k2 < Cin) ? *(const v4i*)(ib0 + poff + k2) : (v4i)0;
    v4i b10 = (k1 < Cin) ? *(const v4i*)(ib1 + poff + k1) : (v4i)0;
    v4i b11 = (k2 < Cin) ? *(const v4i*)(ib1 + poff + k2) : (v4i)0;

    v8i B0, B1;
    B0[0] = b00.x; B0[1] = b00.y; B0[2] = b00.z; B0[3] = b00.w;
    B0[4] = b01.x; B0[5] = b01.y; B0[6] = b01.z; B0[7] = b01.w;
    B1[0] = b10.x; B1[1] = b10.y; B1[2] = b10.z; B1[3] = b10.w;
    B1[4] = b11.x; B1[5] = b11.y; B1[6] = b11.z; B1[7] = b11.w;

    acc0 = __builtin_amdgcn_wmma_i32_16x16x64_iu8(true, A, true, B0, acc0,
                                                  false, false);
    acc1 = __builtin_amdgcn_wmma_i32_16x16x64_iu8(true, A, true, B1, acc1,
                                                  false, false);
  }

  i8* ob0 = out + (size_t)((n0 * Ho + y0 + OPAD) * Wo + x0 + OPAD) * Cout;
  i8* ob1 = out + (size_t)((n1 * Ho + y1 + OPAD) * Wo + x1 + OPAD) * Cout;
#pragma unroll
  for (int r = 0; r < 8; ++r) {
    int co = mt * 16 + r + (hi ? 8 : 0);
    float s = sc[co], t = tt[co];
    ob0[co] = binact_q((float)acc0[r] * s + t);
    ob1[co] = binact_q((float)acc1[r] * s + t);
  }
}

template <int KReal, int KPad, int NB>
__global__ void __launch_bounds__(256)
k_bfc_wmma(const i8* __restrict__ act, const i8* __restrict__ wp,
           const float* __restrict__ sc, const float* __restrict__ tt,
           i8* __restrict__ outc, float* __restrict__ outf,
           int Nf, int nfgroups) {
  constexpr int MT     = BATCH / 16;
  constexpr int CHUNKS = KPad / 64;

  int wave = (blockIdx.x * 256 + threadIdx.x) >> 5;
  int lane = threadIdx.x & 31;
  if (wave >= MT * nfgroups) return;

  int fg = wave % nfgroups;
  int bt = wave / nfgroups;
  bool hi  = lane >= 16;
  int  l15 = lane & 15;

  const i8* arow = act + (size_t)(bt * 16 + l15) * KReal;
  const int aoff = hi ? 16 : 0;
  const int boff = hi ? 16 : 0;

  const i8* wrow[NB];
  v8i acc[NB];
#pragma unroll
  for (int j = 0; j < NB; ++j) {
    wrow[j] = wp + (size_t)((fg * NB + j) * 16 + l15) * KPad;
    acc[j] = (v8i)0;
  }

#pragma unroll
  for (int c = 0; c < CHUNKS; ++c) {
    const int k0 = c * 64;
    v8i A;
#pragma unroll
    for (int j = 0; j < 4; ++j) {
      const int k = k0 + (j & 1) * 8 + (j >> 1) * 32;
      v2i a = (k + aoff < KReal) ? *(const v2i*)(arow + k + aoff) : (v2i)0;
      A[2 * j]     = a.x;
      A[2 * j + 1] = a.y;
    }
#pragma unroll
    for (int j = 0; j < NB; ++j) {
      v4i b0 = *(const v4i*)(wrow[j] + k0 + boff);
      v4i b1 = *(const v4i*)(wrow[j] + k0 + boff + 32);
      v8i B;
      B[0] = b0.x; B[1] = b0.y; B[2] = b0.z; B[3] = b0.w;
      B[4] = b1.x; B[5] = b1.y; B[6] = b1.z; B[7] = b1.w;
      acc[j] = __builtin_amdgcn_wmma_i32_16x16x64_iu8(true, A, true, B, acc[j],
                                                      false, false);
    }
  }

#pragma unroll
  for (int j = 0; j < NB; ++j) {
    int of = (fg * NB + j) * 16 + l15;
#pragma unroll
    for (int r = 0; r < 8; ++r) {
      int nn = bt * 16 + r + (hi ? 8 : 0);
      float yv = (float)acc[j][r];
      if (sc) yv = yv * sc[of] + tt[of];
      if (outc) {
        outc[(size_t)nn * Nf + of] = binact_q(yv);
      } else if (of < Nf) {
        outf[(size_t)nn * Nf + of] = yv;
      }
    }
  }
}

__global__ void k_pool_nhwc(const i8* __restrict__ in, i8* __restrict__ out,
                            int C, int Hin, int Win, int Hout, int Wout,
                            int opad) {
  int gid = blockIdx.x * 256 + threadIdx.x;
  int total = BATCH * Hout * Wout * C;
  if (gid >= total) return;
  int c  = gid % C;
  int p  = gid / C;
  int xo = p % Wout;
  int t  = p / Wout;
  int yo = t % Hout;
  int nb = t / Hout;
  size_t base = (((size_t)(nb * Hin + yo * 2)) * Win + xo * 2) * C + c;
  i8 r = in[base];
  r = i8max(r, in[base + C]);
  r = i8max(r, in[base + (size_t)Win * C]);
  r = i8max(r, in[base + (size_t)Win * C + C]);
  int Hop = Hout + 2 * opad, Wop = Wout + 2 * opad;
  out[(((size_t)(nb * Hop + yo + opad)) * Wop + xo + opad) * C + c] = r;
}

__global__ void k_pool5_nchw(const i8* __restrict__ in, i8* __restrict__ out) {
  const int C = 96;
  int gid = blockIdx.x * 256 + threadIdx.x;
  int total = BATCH * 3 * 3 * C;
  if (gid >= total) return;
  int c  = gid % C;
  int p  = gid / C;
  int xo = p % 3;
  int t  = p / 3;
  int yo = t % 3;
  int nb = t / 3;
  size_t base = (((size_t)(nb * 7 + yo * 2)) * 7 + xo * 2) * C + c;
  i8 r = in[base];
  r = i8max(r, in[base + C]);
  r = i8max(r, in[base + 7 * C]);
  r = i8max(r, in[base + 7 * C + C]);
  out[(size_t)nb * 864 + c * 9 + yo * 3 + xo] = r;
}

static inline int cdiv(long long a, long long b) { return (int)((a + b - 1) / b); }

extern "C" void kernel_launch(void* const* d_in, const int* in_sizes, int n_in,
                              void* d_out, int out_size, void* d_ws, size_t ws_size,
                              hipStream_t stream) {
  (void)in_sizes; (void)n_in; (void)out_size; (void)ws_size;

  const float* x   = (const float*)d_in[0];
  const float* w1  = (const float*)d_in[1];
  const float* w2  = (const float*)d_in[2];
  const float* w3  = (const float*)d_in[3];
  const float* w4  = (const float*)d_in[4];
  const float* w5  = (const float*)d_in[5];
  const float *g1 = (const float*)d_in[6],  *b1 = (const float*)d_in[7],
              *m1 = (const float*)d_in[8],  *v1 = (const float*)d_in[9];
  const float *g2 = (const float*)d_in[10], *b2 = (const float*)d_in[11],
              *m2 = (const float*)d_in[12], *v2 = (const float*)d_in[13];
  const float *g3 = (const float*)d_in[14], *b3 = (const float*)d_in[15],
              *m3 = (const float*)d_in[16], *v3 = (const float*)d_in[17];
  const float *g4 = (const float*)d_in[18], *b4 = (const float*)d_in[19],
              *m4 = (const float*)d_in[20], *v4 = (const float*)d_in[21];
  const float *g5 = (const float*)d_in[22], *b5 = (const float*)d_in[23],
              *m5 = (const float*)d_in[24], *v5 = (const float*)d_in[25];
  const float* fw1 = (const float*)d_in[26];
  const float* fb1 = (const float*)d_in[27];
  const float* fw2 = (const float*)d_in[28];
  const float* fb2 = (const float*)d_in[29];
  const float* fw3 = (const float*)d_in[30];
  const float *fg1 = (const float*)d_in[31], *fbb1 = (const float*)d_in[32],
              *fm1 = (const float*)d_in[33], *fv1  = (const float*)d_in[34];
  const float *fg2 = (const float*)d_in[35], *fbb2 = (const float*)d_in[36],
              *fm2 = (const float*)d_in[37], *fv2  = (const float*)d_in[38];
  float* out = (float*)d_out;

  const size_t SZ_A1P = (size_t)BATCH * 30 * 30 * 96;
  const size_t SZ_C2O = (size_t)BATCH * 28 * 28 * 96;
  const size_t SZ_A2P = (size_t)BATCH * 16 * 16 * 96;
  const size_t SZ_A3P = (size_t)BATCH * 9 * 9 * 128;
  char* ws = (char*)d_ws;
  i8* R1 = (i8*)(ws);
  i8* R2 = (i8*)(ws + SZ_A1P);
  i8* R3 = (i8*)(ws + SZ_A1P + SZ_C2O);
  char* Wb = ws + SZ_A1P + SZ_C2O + SZ_A2P;
  i8* w2p  = (i8*)(Wb + 0);
  i8* w3p  = (i8*)(Wb + 110592);
  i8* w4p  = (i8*)(Wb + 258048);
  i8* w5p  = (i8*)(Wb + 405504);
  i8* fw1p = (i8*)(Wb + 516096);
  i8* fw2p = (i8*)(Wb + 974848);
  i8* fw3p = (i8*)(Wb + 1007616);
  float* bn = (float*)(Wb + 1008640);
  float *s1 = bn + 0,    *t1 = bn + 512;
  float *s2 = bn + 1024, *t2 = bn + 1536;
  float *s3 = bn + 2048, *t3 = bn + 2560;
  float *s4 = bn + 3072, *t4 = bn + 3584;
  float *s5 = bn + 4096, *t5 = bn + 4608;
  float *fs1 = bn + 5120, *ft1 = bn + 5632;
  float *fs2 = bn + 6144, *ft2 = bn + 6656;

  hipMemsetAsync(R1, 0, SZ_A1P, stream);
  hipMemsetAsync(R3, 0, SZ_A2P, stream);

  k_prep_wconv<<<cdiv(96  * 1152, 256), 256, 0, stream>>>(w2, w2p, 96, 96, 1152);
  k_prep_wconv<<<cdiv(128 * 1152, 256), 256, 0, stream>>>(w3, w3p, 128, 96, 1152);
  k_prep_wconv<<<cdiv(128 * 1152, 256), 256, 0, stream>>>(w4, w4p, 128, 128, 1152);
  k_prep_wconv<<<cdiv(96  * 1152, 256), 256, 0, stream>>>(w5, w5p, 96, 128, 1152);
  k_prep_wfc<<<cdiv(512 * 896, 256), 256, 0, stream>>>(fw1, fw1p, 512, 512, 896, 864);
  k_prep_wfc<<<cdiv(64  * 512, 256), 256, 0, stream>>>(fw2, fw2p, 64, 64, 512, 512);
  k_prep_wfc<<<cdiv(16  * 64,  256), 256, 0, stream>>>(fw3, fw3p, 16, 10, 64, 64);
  k_prep_bn<<<1, 256, 0, stream>>>(g1, b1, m1, v1, nullptr, 96,  s1, t1);
  k_prep_bn<<<1, 256, 0, stream>>>(g2, b2, m2, v2, nullptr, 96,  s2, t2);
  k_prep_bn<<<1, 256, 0, stream>>>(g3, b3, m3, v3, nullptr, 128, s3, t3);
  k_prep_bn<<<1, 256, 0, stream>>>(g4, b4, m4, v4, nullptr, 128, s4, t4);
  k_prep_bn<<<1, 256, 0, stream>>>(g5, b5, m5, v5, nullptr, 96,  s5, t5);
  k_prep_bn<<<2, 256, 0, stream>>>(fg1, fbb1, fm1, fv1, fb1, 512, fs1, ft1);
  k_prep_bn<<<1, 256, 0, stream>>>(fg2, fbb2, fm2, fv2, fb2, 64,  fs2, ft2);

  k_conv1<<<cdiv((long long)BATCH * 28 * 28 * 96, 256), 256, 0, stream>>>(
      x, w1, s1, t1, R1);

  {
    long long waves = (96 / 16) * ((long long)BATCH * 28 * 28 / 32);
    k_bconv_wmma<96, 28, 28, 96, 0><<<cdiv(waves * 32, 256), 256, 0, stream>>>(
        R1, w2p, s2, t2, R2);
  }
  k_pool_nhwc<<<cdiv((long long)BATCH * 14 * 14 * 96, 256), 256, 0, stream>>>(
      R2, R3, 96, 28, 28, 14, 14, 1);

  {
    long long waves = (128 / 16) * ((long long)BATCH * 14 * 14 / 32);
    k_bconv_wmma<96, 14, 14, 128, 0><<<cdiv(waves * 32, 256), 256, 0, stream>>>(
        R3, w3p, s3, t3, R1);
  }
  hipMemsetAsync(R2, 0, SZ_A3P, stream);
  k_pool_nhwc<<<cdiv((long long)BATCH * 7 * 7 * 128, 256), 256, 0, stream>>>(
      R1, R2, 128, 14, 14, 7, 7, 1);

  hipMemsetAsync(R3, 0, SZ_A3P, stream);
  {
    long long waves = (128 / 16) * ((long long)BATCH * 7 * 7 / 32);
    k_bconv_wmma<128, 7, 7, 128, 1><<<cdiv(waves * 32, 256), 256, 0, stream>>>(
        R2, w4p, s4, t4, R3);
  }

  {
    long long waves = (96 / 16) * ((long long)BATCH * 7 * 7 / 32);
    k_bconv_wmma<128, 7, 7, 96, 0><<<cdiv(waves * 32, 256), 256, 0, stream>>>(
        R3, w5p, s5, t5, R1);
  }
  k_pool5_nchw<<<cdiv((long long)BATCH * 3 * 3 * 96, 256), 256, 0, stream>>>(R1, R2);

  {
    long long waves = (BATCH / 16) * 16;
    k_bfc_wmma<864, 896, 2><<<cdiv(waves * 32, 256), 256, 0, stream>>>(
        R2, fw1p, fs1, ft1, R1, nullptr, 512, 16);
  }
  {
    long long waves = (BATCH / 16) * 2;
    k_bfc_wmma<512, 512, 2><<<cdiv(waves * 32, 256), 256, 0, stream>>>(
        R1, fw2p, fs2, ft2, R3, nullptr, 64, 2);
  }
  {
    long long waves = (BATCH / 16) * 1;
    k_bfc_wmma<64, 64, 1><<<cdiv(waves * 32, 256), 256, 0, stream>>>(
        R3, fw3p, nullptr, nullptr, nullptr, out, 10, 1);
  }
}
